// RNN_model_81020263072364
// MI455X (gfx1250) — compile-verified, matrix operand order checked by reading, not run on hardware
//
#include <hip/hip_runtime.h>
#include <hip/hip_bf16.h>
#include <math.h>

// Problem dims (fixed by reference)
#define T_SEQ 2048
#define E_DIM 512
#define H_DIM 1024
#define G_DIM 4096   // 4*H
#define V_DIM 32000

// LDS tile geometry for the WMMA GEMM
#define BM 128            // block tile M
#define BN 128            // block tile N
#define BK 32             // K slab
#define LDS_STRIDE 40     // elements per LDS row (32 data + 8 pad -> 80B, kills bank conflicts)

typedef __bf16 v16bf __attribute__((ext_vector_type(16)));
typedef float  v8f   __attribute__((ext_vector_type(8)));

// ---------- helpers ----------
__device__ __forceinline__ __bf16 f2bf(float f) {
  union { float f; unsigned u; } a; a.f = f;
  unsigned u = a.u;
  u += 0x7FFFu + ((u >> 16) & 1u);            // round-to-nearest-even
  union { unsigned short s; __bf16 b; } o; o.s = (unsigned short)(u >> 16);
  return o.b;
}

// 16x32 bf16 fragment from LDS (A-matrix layout per CDNA5 ISA 7.12.2):
//   lanes 0-15 : row = lane,    K = {0..7, 16..23}
//   lanes 16-31: row = lane-16, K = {8..15, 24..31}
// base points at (tile_row0, k=0) inside an LDS tile with row stride LDS_STRIDE.
__device__ __forceinline__ v16bf lds_frag16x32(const __bf16* base, int lane) {
  const int r = lane & 15, h = lane >> 4;
  const __bf16* q = base + r * LDS_STRIDE + h * 8;
  v16bf v;
#pragma unroll
  for (int i = 0; i < 8; ++i) v[i] = q[i];
#pragma unroll
  for (int i = 0; i < 8; ++i) v[8 + i] = q[16 + i];
  return v;
}

// Issue async global->LDS copies for one 128x32 bf16 tile (8KB payload).
// 512 chunks of 16B; 256 threads issue 2 GLOBAL_LOAD_ASYNC_TO_LDS_B128 each
// (tracked by ASYNCcnt; in-order completion per wave).
__device__ __forceinline__ void async_copy_tile(const __bf16* gbase, int ldg,
                                                unsigned lds_base, int tid) {
#pragma unroll
  for (int it = 0; it < 2; ++it) {
    const int slot = tid + it * 256;
    const int row = slot >> 2;          // 0..127
    const int seg = slot & 3;           // 16B segment within the 64B row
    const unsigned long long g =
        (unsigned long long)(uintptr_t)(gbase + (size_t)row * ldg) + (unsigned)(seg * 16);
    const unsigned l = lds_base + (unsigned)(row * (LDS_STRIDE * 2) + seg * 16);
    asm volatile("global_load_async_to_lds_b128 %0, %1, off"
                 :: "v"(l), "v"(g) : "memory");
  }
}

// ---------- prep kernels ----------
// dst (cols x rows, bf16) = transpose of src (rows x cols, f32)
__global__ void k_transpose_to_bf16(const float* __restrict__ src, __bf16* __restrict__ dst,
                                    int rows, int cols) {
  size_t gid = (size_t)blockIdx.x * blockDim.x + threadIdx.x;
  size_t total = (size_t)rows * cols;
  if (gid >= total) return;
  int c = (int)(gid / rows);
  int r = (int)(gid % rows);
  dst[gid] = f2bf(src[(size_t)r * cols + c]);
}

// dst (cols x rows, f32) = transpose of src (rows x cols, f32)
__global__ void k_transpose_f32(const float* __restrict__ src, float* __restrict__ dst,
                                int rows, int cols) {
  size_t gid = (size_t)blockIdx.x * blockDim.x + threadIdx.x;
  size_t total = (size_t)rows * cols;
  if (gid >= total) return;
  int c = (int)(gid / rows);
  int r = (int)(gid % rows);
  dst[gid] = src[(size_t)r * cols + c];
}

__global__ void k_cvt_bf16(const float* __restrict__ src, __bf16* __restrict__ dst, size_t n) {
  size_t gid = (size_t)blockIdx.x * blockDim.x + threadIdx.x;
  if (gid < n) dst[gid] = f2bf(src[gid]);
}

// Embedding gather -> bf16 activations
__global__ void k_embed(const int* __restrict__ sent, const float* __restrict__ emb,
                        __bf16* __restrict__ x) {
  size_t gid = (size_t)blockIdx.x * blockDim.x + threadIdx.x;
  if (gid >= (size_t)T_SEQ * E_DIM) return;
  int t = (int)(gid >> 9);
  int e = (int)(gid & 511);
  int tok = sent[t];
  x[gid] = f2bf(emb[(size_t)tok * E_DIM + e]);
}

// bias = bi + bh; zero h, c; zero grid-barrier counters (fresh each launch ->
// deterministic across graph replays)
__global__ void k_init(const float* __restrict__ bi, const float* __restrict__ bh,
                       float* __restrict__ bias, float* __restrict__ h, float* __restrict__ c,
                       int* __restrict__ bar) {
  int gid = blockIdx.x * blockDim.x + threadIdx.x;
  if (gid < G_DIM) bias[gid] = bi[gid] + bh[gid];
  if (gid < H_DIM) { h[gid] = 0.0f; c[gid] = 0.0f; }
  if (gid < 2) bar[gid] = 0;
}

// ---------- LDS-staged WMMA GEMM ----------
// C(MxN,f32) = A(MxK,bf16) @ BT(NxK,bf16)^T + biasN [+ relu]
// 256 threads = 8 wave32s (2M x 4N). Block tile 128x128, K slab 32,
// double-buffered async global->LDS staging. Each wave: 64x32 via 4x2 WMMA.
__global__ void wmma_gemm_bf16(const __bf16* __restrict__ A, const __bf16* __restrict__ BT,
                               const float* __restrict__ biasN, float* __restrict__ C,
                               int M, int N, int K, int relu) {
  __shared__ __bf16 As[2][BM * LDS_STRIDE];
  __shared__ __bf16 Bs[2][BM * LDS_STRIDE];

  const int tid  = threadIdx.x;
  const int lane = tid & 31;
  const int wave = tid >> 5;
  const int wm = wave & 1;          // 0..1 -> 64-row group
  const int wn = wave >> 1;         // 0..3 -> 32-col group
  const int rowB = blockIdx.y * BM;
  const int colB = blockIdx.x * BN;

  const unsigned lA[2] = { (unsigned)(uintptr_t)&As[0][0], (unsigned)(uintptr_t)&As[1][0] };
  const unsigned lB[2] = { (unsigned)(uintptr_t)&Bs[0][0], (unsigned)(uintptr_t)&Bs[1][0] };

  const __bf16* Ablk = A  + (size_t)rowB * K;
  const __bf16* Bblk = BT + (size_t)colB * K;

  // prologue: fill buffer 0 (4 async b128 per thread outstanding)
  async_copy_tile(Ablk, K, lA[0], tid);
  async_copy_tile(Bblk, K, lB[0], tid);

  v8f z = {};
  v8f acc[4][2];
#pragma unroll
  for (int i = 0; i < 4; ++i)
#pragma unroll
    for (int j = 0; j < 2; ++j) acc[i][j] = z;

  const int nslab = K / BK;
  for (int s = 0; s < nslab; ++s) {
    const int cur = s & 1, nxt = cur ^ 1;
    if (s + 1 < nslab) {
      // issue next slab, then retire the 4 in-order loads of the current one
      async_copy_tile(Ablk + (s + 1) * BK, K, lA[nxt], tid);
      async_copy_tile(Bblk + (s + 1) * BK, K, lB[nxt], tid);
      asm volatile("s_wait_asynccnt 0x4" ::: "memory");
    } else {
      asm volatile("s_wait_asynccnt 0x0" ::: "memory");
    }
    __syncthreads();   // all waves' async fills of `cur` are complete & visible

    const __bf16* a_base = &As[cur][0] + (wm * 64) * LDS_STRIDE;
    const __bf16* b_base = &Bs[cur][0] + (wn * 32) * LDS_STRIDE;

    v16bf af[4], bfrag[2];
#pragma unroll
    for (int i = 0; i < 4; ++i) af[i]    = lds_frag16x32(a_base + i * 16 * LDS_STRIDE, lane);
#pragma unroll
    for (int j = 0; j < 2; ++j) bfrag[j] = lds_frag16x32(b_base + j * 16 * LDS_STRIDE, lane);

#pragma unroll
    for (int i = 0; i < 4; ++i)
#pragma unroll
      for (int j = 0; j < 2; ++j)
        acc[i][j] = __builtin_amdgcn_wmma_f32_16x16x32_bf16(
            false, af[i], false, bfrag[j], (short)0, acc[i][j], false, false);

    __syncthreads();   // protect `cur` from being overwritten by the s+2 issue
  }

  // Epilogue. C/D layout: VGPR r -> M = r (lanes 0-15) or 8+r (lanes 16-31); N = lane&15.
  const int n_in = lane & 15;
  const int mofs = (lane >> 4) * 8;
#pragma unroll
  for (int i = 0; i < 4; ++i)
#pragma unroll
    for (int j = 0; j < 2; ++j) {
      const int col = colB + wn * 32 + 16 * j + n_in;
      const float b = biasN[col];
#pragma unroll
      for (int vr = 0; vr < 8; ++vr) {
        const int rowm = rowB + wm * 64 + 16 * i + mofs + vr;
        float v = acc[i][j][vr] + b;
        if (relu) v = fmaxf(v, 0.0f);
        C[(size_t)rowm * N + col] = v;
      }
    }
}

// ---------- persistent LSTM scan (grid-synced) ----------
__device__ __forceinline__ void grid_sync(int* bar) {
  __syncthreads();
  if (threadIdx.x == 0) {
    __threadfence();
    volatile int* vgen = bar + 1;
    int gen = *vgen;
    if (atomicAdd(bar, 1) == (int)gridDim.x - 1) {
      bar[0] = 0;
      __threadfence();
      atomicAdd(bar + 1, 1);
    } else {
      while (*vgen == gen) __builtin_amdgcn_s_sleep(1);
    }
    __threadfence();
  }
  __syncthreads();
}

// 16 blocks x 256 threads = 4096 threads: thread gid owns gate lane gid.
__global__ void k_lstm(const float* __restrict__ xg, const float* __restrict__ WhT,
                       float* h, float* c, float* gates, __bf16* __restrict__ hs, int* bar) {
  const int gid = blockIdx.x * blockDim.x + threadIdx.x;   // 0..4095
  const float* wrow = WhT + (size_t)gid * H_DIM;
  for (int t = 0; t < T_SEQ; ++t) {
    float acc = xg[(size_t)t * G_DIM + gid];
#pragma unroll 4
    for (int k = 0; k < H_DIM; k += 4) {
      const float4 hv = *(const float4*)(h + k);
      const float4 wv = *(const float4*)(wrow + k);
      acc = fmaf(hv.x, wv.x, acc);
      acc = fmaf(hv.y, wv.y, acc);
      acc = fmaf(hv.z, wv.z, acc);
      acc = fmaf(hv.w, wv.w, acc);
    }
    gates[gid] = acc;
    grid_sync(bar);
    if (gid < H_DIM) {
      const float ig = 1.0f / (1.0f + __expf(-gates[gid]));
      const float fg = 1.0f / (1.0f + __expf(-gates[H_DIM + gid]));
      const float cg = tanhf(gates[2 * H_DIM + gid]);
      const float og = 1.0f / (1.0f + __expf(-gates[3 * H_DIM + gid]));
      const float cn = fg * c[gid] + ig * cg;
      const float hn = og * tanhf(cn);
      c[gid] = cn;
      h[gid] = hn;
      hs[(size_t)t * H_DIM + gid] = f2bf(hn);
    }
    grid_sync(bar);
  }
}

// ---------- in-place log-softmax over V per row ----------
__global__ void k_logsoftmax(float* __restrict__ out) {
  float* row = out + (size_t)blockIdx.x * V_DIM;
  __shared__ float sm[256], ss[256];
  const int tid = threadIdx.x;
  float m = -3.0e38f, s = 0.0f;
  for (int v = tid; v < V_DIM; v += 256) {
    float x = row[v];
    float nm = fmaxf(m, x);
    s = s * __expf(m - nm) + __expf(x - nm);
    m = nm;
  }
  sm[tid] = m; ss[tid] = s;
  __syncthreads();
  for (int st = 128; st > 0; st >>= 1) {
    if (tid < st) {
      float m2 = fmaxf(sm[tid], sm[tid + st]);
      float s2 = ss[tid] * __expf(sm[tid] - m2) + ss[tid + st] * __expf(sm[tid + st] - m2);
      sm[tid] = m2; ss[tid] = s2;
    }
    __syncthreads();
  }
  const float lse = sm[0] + __logf(ss[0]);
  for (int v = tid; v < V_DIM; v += 256) row[v] -= lse;
}

// ---------- host orchestration ----------
extern "C" void kernel_launch(void* const* d_in, const int* in_sizes, int n_in,
                              void* d_out, int out_size, void* d_ws, size_t ws_size,
                              hipStream_t stream) {
  (void)in_sizes; (void)n_in; (void)out_size; (void)ws_size;
  const int*   sent = (const int*)  d_in[0];
  const float* emb  = (const float*)d_in[1];
  const float* Wi   = (const float*)d_in[2];
  const float* Wh   = (const float*)d_in[3];
  const float* bi   = (const float*)d_in[4];
  const float* bh   = (const float*)d_in[5];
  const float* fcw  = (const float*)d_in[6];
  const float* fcb  = (const float*)d_in[7];
  float* out = (float*)d_out;

  char* ws = (char*)d_ws;
  size_t off = 0;
  auto carve = [&](size_t bytes) -> char* {
    char* p = ws + off;
    off = (off + bytes + 255) & ~(size_t)255;
    return p;
  };
  __bf16* WiT  = (__bf16*)carve((size_t)G_DIM * E_DIM * 2);   // Wi^T  (4096 x 512)  bf16
  __bf16* FcwB = (__bf16*)carve((size_t)V_DIM * H_DIM * 2);   // fc_w  (32000x1024)  bf16
  __bf16* Xb   = (__bf16*)carve((size_t)T_SEQ * E_DIM * 2);   // gathered embeddings bf16
  float*  Xg   = (float*) carve((size_t)T_SEQ * G_DIM * 4);   // x@Wi + bi + bh
  float*  WhT  = (float*) carve((size_t)G_DIM * H_DIM * 4);   // Wh^T  (4096 x 1024) f32
  __bf16* Hs   = (__bf16*)carve((size_t)T_SEQ * H_DIM * 2);   // LSTM outputs bf16
  float*  Hst  = (float*) carve(H_DIM * 4);
  float*  Cst  = (float*) carve(H_DIM * 4);
  float*  Gt   = (float*) carve(G_DIM * 4);
  float*  Bias = (float*) carve(G_DIM * 4);
  int*    Bar  = (int*)   carve(256);

  dim3 blk(256);
  k_transpose_to_bf16<<<dim3(((size_t)G_DIM * E_DIM + 255) / 256), blk, 0, stream>>>(Wi, WiT, E_DIM, G_DIM);
  k_transpose_f32   <<<dim3(((size_t)G_DIM * H_DIM + 255) / 256), blk, 0, stream>>>(Wh, WhT, H_DIM, G_DIM);
  k_cvt_bf16        <<<dim3(((size_t)V_DIM * H_DIM + 255) / 256), blk, 0, stream>>>(fcw, FcwB, (size_t)V_DIM * H_DIM);
  k_embed           <<<dim3(((size_t)T_SEQ * E_DIM + 255) / 256), blk, 0, stream>>>(sent, emb, Xb);
  k_init            <<<dim3(16), blk, 0, stream>>>(bi, bh, Bias, Hst, Cst, Bar);

  // xg = x @ Wi + (bi+bh) : 2048 x 4096, K=512
  wmma_gemm_bf16<<<dim3(G_DIM / BN, T_SEQ / BM), blk, 0, stream>>>(Xb, WiT, Bias, Xg, T_SEQ, G_DIM, E_DIM, 0);

  // sequential recurrence
  k_lstm<<<dim3(16), blk, 0, stream>>>(Xg, WhT, Hst, Cst, Gt, Hs, Bar);

  // logits = relu(hs @ fc_w^T + fc_b) : 2048 x 32000, K=1024 -> d_out
  wmma_gemm_bf16<<<dim3(V_DIM / BN, T_SEQ / BM), blk, 0, stream>>>(Hs, FcwB, fcb, out, T_SEQ, V_DIM, H_DIM, 1);

  k_logsoftmax<<<dim3(T_SEQ), blk, 0, stream>>>(out);
}
